// FourierAttention_66357244723219
// MI455X (gfx1250) — matrix operand order repaired and checked by reading, not compiled, not run on hardware
//
#include <hip/hip_runtime.h>

// Fourier attention, fused for MI455X (gfx1250, wave32).
// B=4 S=1024 D=768 H=12 WD=64.
// Phase 1: QKV projection via v_wmma_f32_16x16x32_bf16 (bf16 in, fp32 acc).
//          Q,K -> fp32 [B,H,S,64]; V -> bf16 transposed [B,H,64,S].
// Phase 2: fused sinc^4-product scores (fp32 VALU; ~3.2G v_sin_f32 dominates)
//          + unnormalized softmax P.V via WMMA (4 PV tiles + 1 ones-tile for
//          the row-sum denominator) per 32-key block.
//          K blocks (8 KB contiguous tiles) and per-wave q tiles (4 KB) are
//          DMA'd into LDS by the Tensor Data Mover (tensor_load_to_lds),
//          double-buffered (K) and synchronized with s_wait_tensorcnt + one
//          workgroup barrier per block.

#define Bb 4
#define Ss 1024
#define Dd 768
#define Hh 12
#define WDw 64

typedef __attribute__((ext_vector_type(16))) __bf16    v16bf;
typedef __attribute__((ext_vector_type(8)))  float     v8f;
typedef __attribute__((ext_vector_type(4)))  unsigned  v4u;
typedef __attribute__((ext_vector_type(8)))  int       v8i;
typedef __attribute__((ext_vector_type(4)))  int       v4i;

#if __has_builtin(__builtin_amdgcn_tensor_load_to_lds)
#define HAVE_TDM 1
#else
#define HAVE_TDM 0
#endif

__device__ __forceinline__ float sinc4(float d) {
  const float PI_F = 3.14159265358979323846f;
  float t = d * PI_F;
  float s = __fdividef(__sinf(t), t);     // v_sin_f32 + fast divide
  s = (fabsf(d) < 1e-9f) ? 1.0f : s;      // sinc(0) = 1
  float s2 = s * s;
  return s2 * s2;                         // sinc^4
}

// K index held by A-fragment element e for a lane in half `hf`:
// contiguous order, K = hf*16 + e (matches the V B-fragment loads).
__device__ __forceinline__ int a_k_of_elem(int e, int hf) {
  return hf * 16 + e;
}

#if HAVE_TDM
// TDM: DMA one contiguous rows x 64 fp32 tile from global into LDS.
// D# group0: count=1 | lds_addr | global_addr(57b) | type=2.
// D# group1: data_size=4B; tensor_dim0=64, stride0=64; tensor_dim1=1024;
//            tile_dim0=64, tile_dim1=rows (2D tile; dims 2+ unused).
__device__ __forceinline__ void tdm_load_rows(const float* gsrc, float* lds_dst,
                                              int rows) {
  unsigned glo = (unsigned)(uintptr_t)gsrc;
  unsigned ghi = (unsigned)((uintptr_t)gsrc >> 32);
  v4u g0;
  g0[0] = 1u;                                        // count=1 (valid), no gather
  g0[1] = (unsigned)(uintptr_t)lds_dst;              // LDS byte address
  g0[2] = glo;                                       // global_addr[31:0]
  g0[3] = (ghi & 0x01FFFFFFu) | (2u << 30);          // global_addr[56:32] | type=2
  v8i g1;
  g1[0] = 2 << 16;                                   // data_size = 4 bytes
  g1[1] = (WDw & 0xFFFF) << 16;                      // tensor_dim0[15:0] = 64
  g1[2] = (WDw >> 16) | ((Ss & 0xFFFF) << 16);       // dim0 hi | tensor_dim1[15:0]
  g1[3] = (Ss >> 16) | (WDw << 16);                  // dim1 hi | tile_dim0 = 64
  g1[4] = rows;                                      // tile_dim1, tile_dim2 = 0
  g1[5] = WDw;                                       // tensor_dim0_stride = 64
  g1[6] = 0;                                         // stride0 hi | stride1 lo
  g1[7] = 0;                                         // stride1 hi
  v4i z4 = {0, 0, 0, 0};
#if __clang_major__ >= 23
  v8i z8 = {0, 0, 0, 0, 0, 0, 0, 0};
  __builtin_amdgcn_tensor_load_to_lds(g0, g1, z4, z4, z8, 0);
#else
  __builtin_amdgcn_tensor_load_to_lds(g0, g1, z4, z4, 0);
#endif
}
#endif  // HAVE_TDM

// ---------------------------------------------------------------------------
// Kernel 2 (placed first so the disassembly window shows it): fused attention.
// grid (8 q-groups, 48 (b,h)), block 256 = 8 waves, one 16-row q tile/wave.
// K blocks double-buffered via TDM; q tiles TDM'd per wave.
// ---------------------------------------------------------------------------
__global__ __launch_bounds__(256) void fourier_attn_kernel(
    const float* __restrict__ q_ws, const float* __restrict__ k_ws,
    const __bf16* __restrict__ vT, const int* __restrict__ mask,
    float* __restrict__ out) {
  __shared__ float k_lds[2][32 * WDw];        // 16 KB double-buffered K blocks
  __shared__ float q_lds[8][16 * WDw];        // 32 KB per-wave q tiles
  __shared__ float madj[Ss];                  // 4 KB: 10000*(1-mask[b, :])

  const int tid  = threadIdx.x;
  const int lane = tid & 31;
  const int wave = tid >> 5;
  const int hf   = lane >> 4;
  const int m    = lane & 15;

  const int bh = blockIdx.y;
  const int b  = bh / Hh;
  const int h  = bh % Hh;
  const int qbase = (blockIdx.x * 8 + wave) * 16;

  const float* qsrc    = q_ws + ((size_t)bh * Ss + qbase) * WDw;
  const float* kbase_g = k_ws + (size_t)bh * Ss * WDw;

#if HAVE_TDM
  // Each wave DMAs its own contiguous 16x64 fp32 q tile (TDM issues once per
  // wave, EXEC ignored); wave 0 additionally DMAs K block 0.
  tdm_load_rows(qsrc, &q_lds[wave][0], 16);
  if (wave == 0) tdm_load_rows(kbase_g, &k_lds[0][0], 32);
#else
  {
#pragma unroll
    for (int i = 0; i < 8; ++i) {
      int idx = lane * 32 + i * 4;
      *(float4*)&q_lds[wave][idx] = *(const float4*)(qsrc + idx);
    }
    const int idx = tid * 8;
    *(float4*)&k_lds[0][idx]     = *(const float4*)(kbase_g + idx);
    *(float4*)&k_lds[0][idx + 4] = *(const float4*)(kbase_g + idx + 4);
  }
#endif

  {  // stage mask adjustment for all S keys of this batch row, once
#pragma unroll
    for (int j = 0; j < 4; ++j) {
      int idx = tid * 4 + j;
      madj[idx] = 10000.0f * (1.0f - (float)mask[b * Ss + idx]);
    }
  }

#if HAVE_TDM
  // Wait for this wave's own q DMA (and, for wave 0, the K block 0 DMA).
  __builtin_amdgcn_s_wait_tensorcnt(0);
#endif
  __syncthreads();                            // publish K block 0 + madj

  v8f acc0 = {}, acc1 = {}, acc2 = {}, acc3 = {}, den = {};
  v16bf ones;
#pragma unroll
  for (int e = 0; e < 16; ++e) ones[e] = (__bf16)1.0f;

  const float* qrow = &q_lds[wave][m * WDw];

  for (int kb = 0; kb < 32; ++kb) {
    const int buf = kb & 1;

    // kick off next block's DMA into the other buffer (safe: previous
    // compute on it finished before the barrier that ended iteration kb-1)
#if HAVE_TDM
    if (wave == 0 && kb + 1 < 32)
      tdm_load_rows(kbase_g + (size_t)(kb + 1) * 32 * WDw, &k_lds[buf ^ 1][0], 32);
#endif

    // ---- scores: 16 running sinc^4 products, q loaded once per w4 ----
    float prod[16];
#pragma unroll
    for (int e = 0; e < 16; ++e) prod[e] = 1.0f;
#pragma unroll 1
    for (int w4 = 0; w4 < 16; ++w4) {         // 64 dims, 4 at a time (ds b128)
      const float4 qv = *(const float4*)(qrow + w4 * 4);
#pragma unroll
      for (int e = 0; e < 16; ++e) {
        const int K = a_k_of_elem(e, hf);
        const float4 kv = *(const float4*)(&k_lds[buf][K * WDw + w4 * 4]);
        prod[e] *= sinc4(qv.x - kv.x) * sinc4(qv.y - kv.y) *
                   sinc4(qv.z - kv.z) * sinc4(qv.w - kv.w);
      }
    }
    // scores in [0,1] -> exp() directly is safe; no online max needed.
    v16bf pfrag;
#pragma unroll
    for (int e = 0; e < 16; ++e) {
      const int Kg = kb * 32 + a_k_of_elem(e, hf);
      pfrag[e] = (__bf16)__expf(prod[e] - madj[Kg]);
    }

    // ---- P.V + denominator via WMMA ----
    // B fragments from transposed bf16 V: lane = column w = t*16 + m,
    // K = hf*16 + e -> 16 contiguous bf16 (32B-aligned v16bf loads).
    // reuse_a on the trailing identical WMMAs: same A operand (pfrag).
    const size_t vcol = (size_t)kb * 32 + hf * 16;
    v16bf v0 = *(const v16bf*)(vT + ((size_t)bh * WDw + 0 * 16 + m) * Ss + vcol);
    acc0 = __builtin_amdgcn_wmma_f32_16x16x32_bf16(false, pfrag, false, v0, (short)0, acc0, false, false);
    v16bf v1 = *(const v16bf*)(vT + ((size_t)bh * WDw + 1 * 16 + m) * Ss + vcol);
    acc1 = __builtin_amdgcn_wmma_f32_16x16x32_bf16(false, pfrag, false, v1, (short)0, acc1, true, false);
    v16bf v2 = *(const v16bf*)(vT + ((size_t)bh * WDw + 2 * 16 + m) * Ss + vcol);
    acc2 = __builtin_amdgcn_wmma_f32_16x16x32_bf16(false, pfrag, false, v2, (short)0, acc2, true, false);
    v16bf v3 = *(const v16bf*)(vT + ((size_t)bh * WDw + 3 * 16 + m) * Ss + vcol);
    acc3 = __builtin_amdgcn_wmma_f32_16x16x32_bf16(false, pfrag, false, v3, (short)0, acc3, true, false);
    den  = __builtin_amdgcn_wmma_f32_16x16x32_bf16(false, pfrag, false, ones, (short)0, den, true, false);

    // wait for next block's DMA, then one barrier: (a) publishes the new
    // buffer to all waves, (b) retires this buffer so it can be re-filled.
#if HAVE_TDM
    if (wave == 0) __builtin_amdgcn_s_wait_tensorcnt(0);
    __syncthreads();
#else
    __syncthreads();
    if (kb + 1 < 32) {
      const float* ksrc = kbase_g + (size_t)(kb + 1) * 32 * WDw;
      const int idx = tid * 8;
      *(float4*)&k_lds[buf ^ 1][idx]     = *(const float4*)(ksrc + idx);
      *(float4*)&k_lds[buf ^ 1][idx + 4] = *(const float4*)(ksrc + idx + 4);
    }
    __syncthreads();
#endif
  }

  // y = (P @ V) / rowsum(P); out[B,S,D], d = h*64 + w.
#pragma unroll
  for (int r = 0; r < 8; ++r) {
    const int s = qbase + r + hf * 8;
    const float dinv = __fdividef(1.0f, den[r]);
    float* orow = out + ((size_t)b * Ss + s) * Dd + h * WDw;
    orow[0 * 16 + m] = acc0[r] * dinv;
    orow[1 * 16 + m] = acc1[r] * dinv;
    orow[2 * 16 + m] = acc2[r] * dinv;
    orow[3 * 16 + m] = acc3[r] * dinv;
  }
}

// ---------------------------------------------------------------------------
// Kernel 1: q/k/v = x @ W^T + b   (x: [4096,768] fp32, W: [768,768] fp32)
// grid (256 row-tiles, 6 col-groups, 3 matrices), block 256 = 8 waves.
// Each wave owns one 16x16 output tile; 24 WMMAs over K=768.
// ---------------------------------------------------------------------------
__global__ __launch_bounds__(256) void qkv_proj_kernel(
    const float* __restrict__ x,
    const float* __restrict__ Wq, const float* __restrict__ bq,
    const float* __restrict__ Wk, const float* __restrict__ bk,
    const float* __restrict__ Wv, const float* __restrict__ bv,
    float* __restrict__ q_ws, float* __restrict__ k_ws,
    __bf16* __restrict__ vT_ws) {
  const int lane = threadIdx.x & 31;
  const int wave = threadIdx.x >> 5;
  const int hf   = lane >> 4;
  const int n    = lane & 15;

  const int rt  = blockIdx.x;                 // rows rt*16..+15 of [B*S]
  const int ct  = blockIdx.y * 8 + wave;      // cols ct*16..+15 of D
  const int mat = blockIdx.z;                 // 0=Q 1=K 2=V

  const float* W    = (mat == 0) ? Wq : ((mat == 1) ? Wk : Wv);
  const float* bias = (mat == 0) ? bq : ((mat == 1) ? bk : bv);

  const int arow = rt * 16 + n;               // A row for this lane (m = lane%16)
  const int dcol = ct * 16 + n;               // B column (output feature d)

  v8f acc = {};
  for (int kt = 0; kt < 24; ++kt) {           // K = 768 = 24 * 32
    v16bf a;                                  // A: x[arow, kt*32 + K]
    const float* xrow = x + (size_t)arow * Dd + kt * 32;
#pragma unroll
    for (int i = 0; i < 8; ++i) {
      int kb = hf * 16 + i * 2;
      float2 xv = *(const float2*)(xrow + kb);
      a[2 * i]     = (__bf16)xv.x;
      a[2 * i + 1] = (__bf16)xv.y;
    }
    v16bf bf;                                 // B[k][n] = W[dcol][k] (q = x @ W^T)
    const float* wrow = W + (size_t)dcol * Dd + kt * 32 + hf * 16;
#pragma unroll
    for (int i = 0; i < 4; ++i) {
      float4 wv = *(const float4*)(wrow + i * 4);
      bf[4 * i]     = (__bf16)wv.x;
      bf[4 * i + 1] = (__bf16)wv.y;
      bf[4 * i + 2] = (__bf16)wv.z;
      bf[4 * i + 3] = (__bf16)wv.w;
    }
    acc = __builtin_amdgcn_wmma_f32_16x16x32_bf16(false, a, false, bf,
                                                  (short)0, acc, false, false);
  }

  const float bval = bias[dcol];
  const int h = dcol >> 6;
  const int w = dcol & 63;
#pragma unroll
  for (int r = 0; r < 8; ++r) {
    const int row = rt * 16 + r + hf * 8;     // C/D layout: m = r + 8*half
    const int b   = row >> 10;
    const int s   = row & 1023;
    const float val = acc[r] + bval;
    const size_t bh = (size_t)b * Hh + h;
    if (mat == 0)      q_ws[(bh * Ss + s) * WDw + w] = val;
    else if (mat == 1) k_ws[(bh * Ss + s) * WDw + w] = val;
    else               vT_ws[(bh * WDw + w) * Ss + s] = (__bf16)val;  // transposed
  }
}

extern "C" void kernel_launch(void* const* d_in, const int* in_sizes, int n_in,
                              void* d_out, int out_size, void* d_ws, size_t ws_size,
                              hipStream_t stream) {
  const float* x    = (const float*)d_in[0];
  const int*   mask = (const int*)d_in[1];
  const float* Wq   = (const float*)d_in[2];
  const float* bq   = (const float*)d_in[3];
  const float* Wk   = (const float*)d_in[4];
  const float* bk   = (const float*)d_in[5];
  const float* Wv   = (const float*)d_in[6];
  const float* bv   = (const float*)d_in[7];
  float* out = (float*)d_out;

  char* ws = (char*)d_ws;
  const size_t qkv_elems = (size_t)Bb * Hh * Ss * WDw;          // 3,145,728
  float*  q_ws = (float*)ws;                                     // 12.58 MB
  float*  k_ws = (float*)(ws + qkv_elems * sizeof(float));       // 12.58 MB
  __bf16* vT   = (__bf16*)(ws + 2 * qkv_elems * sizeof(float));  // 6.29 MB

  dim3 g1(256, 6, 3);
  qkv_proj_kernel<<<g1, 256, 0, stream>>>(x, Wq, bq, Wk, bk, Wv, bv,
                                          q_ws, k_ws, vT);

  dim3 g2(8, Bb * Hh, 1);
  fourier_attn_kernel<<<g2, 256, 0, stream>>>(q_ws, k_ws, vT, mask, out);
}
